// RC_63634235458296
// MI455X (gfx1250) — matrix operand order repaired and checked by reading, not compiled, not run on hardware
//
#include <hip/hip_runtime.h>
#include <hip/hip_bf16.h>

typedef __attribute__((ext_vector_type(16))) _Float16 v16h;
typedef __attribute__((ext_vector_type(8)))  _Float16 v8h;
typedef __attribute__((ext_vector_type(8)))  float    v8f;
typedef __attribute__((ext_vector_type(4)))  float    v4f;

#define B_   256
#define T_   250
#define IN_  700
#define H_   128
#define OUT_ 20
#define BT_  (B_*T_)
#define KMAIN_ 672          // (IN_/32)*32

// workspace byte offsets
#define WS_A1M  ((size_t)0)                       // H*H f16        (32 KB)
#define WS_H    ((size_t)32768)                   // BT*H f16       (16.4 MB)
#define WS_XA   (WS_H  + (size_t)BT_*H_*2)        // BT*H f32       (32.8 MB)
#define WS_SSUM (WS_XA + (size_t)BT_*H_*4)        // B*H  f32       (128 KB)

// output float offsets (out, mem_hist, spk_hist, A_norm)
#define O_OUT  ((size_t)0)
#define O_MEM  ((size_t)(B_*OUT_))
#define O_SPK  (O_MEM + (size_t)B_*(T_+1)*H_)
#define O_ANRM (O_SPK + (size_t)B_*(T_+1)*H_)

__device__ __forceinline__ v16h cvt16(v4f a, v4f b, v4f c, v4f d) {
    v16h r;
    r[0]=(_Float16)a[0];  r[1]=(_Float16)a[1];  r[2]=(_Float16)a[2];  r[3]=(_Float16)a[3];
    r[4]=(_Float16)b[0];  r[5]=(_Float16)b[1];  r[6]=(_Float16)b[2];  r[7]=(_Float16)b[3];
    r[8]=(_Float16)c[0];  r[9]=(_Float16)c[1];  r[10]=(_Float16)c[2]; r[11]=(_Float16)c[3];
    r[12]=(_Float16)d[0]; r[13]=(_Float16)d[1]; r[14]=(_Float16)d[2]; r[15]=(_Float16)d[3];
    return r;
}

// ---------------- A1m = A1_w * A1_mask.T  (to f16)  +  A_norm ----------------
__global__ void prep_kernel(const float* __restrict__ A1_w,
                            const int*   __restrict__ A1_mask,
                            _Float16* __restrict__ a1m,
                            float* __restrict__ anorm_out) {
    __shared__ float red[256];
    int tid = threadIdx.x;
    float acc = 0.f;
    for (int idx = tid; idx < H_*H_; idx += 256) {
        int g = idx >> 7, h = idx & (H_-1);
        float v = A1_w[idx] * (float)A1_mask[h*H_ + g];   // mask.T
        a1m[idx] = (_Float16)v;
        acc += fabsf(v);
    }
    red[tid] = acc;
    __syncthreads();
    for (int s = 128; s > 0; s >>= 1) {
        if (tid < s) red[tid] += red[tid + s];
        __syncthreads();
    }
    if (tid == 0) *anorm_out = red[0];
}

// ---------------- GEMM1: h = relu(x @ fc_in_w^T + b)  -> f16 ----------------
// 8 waves/block; wave w owns N-cols [16w,16w+16), 4 M-subtiles of 16 rows.
// Main K loop (21 chunks of 32) is branch-free; single padded tail chunk.
__global__ __launch_bounds__(256) void gemm_in_kernel(
    const float* __restrict__ x, const float* __restrict__ w,
    const float* __restrict__ bias, _Float16* __restrict__ hout) {
    const int lane = threadIdx.x & 31;
    const int half = lane >> 4;
    const int lr   = lane & 15;
    const int n0   = (threadIdx.x >> 5) * 16;
    const long m_base = (long)blockIdx.x * 64;
    const long wrow   = (long)(n0 + lr) * IN_;

    v8f c[4];
    #pragma unroll
    for (int s = 0; s < 4; ++s) c[s] = (v8f){};

    for (int kc = 0; kc < KMAIN_; kc += 32) {
        const float* wr = w + wrow + kc + 8*half;
        v4f q0 = *(const v4f*)(wr);
        v4f q1 = *(const v4f*)(wr + 4);
        v4f q2 = *(const v4f*)(wr + 16);
        v4f q3 = *(const v4f*)(wr + 20);
        v16h bf = cvt16(q0, q1, q2, q3);
        #pragma unroll
        for (int s = 0; s < 4; ++s) {
            const float* xr = x + (m_base + s*16 + lr)*IN_ + kc + 8*half;
            v4f p0 = __builtin_nontemporal_load((const v4f*)(xr));
            v4f p1 = __builtin_nontemporal_load((const v4f*)(xr + 4));
            v4f p2 = __builtin_nontemporal_load((const v4f*)(xr + 16));
            v4f p3 = __builtin_nontemporal_load((const v4f*)(xr + 20));
            __builtin_prefetch(xr + 32, 0, 3);          // next K chunk
            v16h af = cvt16(p0, p1, p2, p3);
            c[s] = __builtin_amdgcn_wmma_f32_16x16x32_f16(
                       false, af, false, bf, (short)0, c[s], false, false);
        }
    }
    // ---- tail chunk kc=672 (K 672..699): every f4 fully valid or fully 0
    {
        const v4f z = (v4f){0.f, 0.f, 0.f, 0.f};
        const float* wr = w + wrow + KMAIN_ + 8*half;
        v4f q0 = *(const v4f*)(wr);
        v4f q1 = *(const v4f*)(wr + 4);
        v4f q2 = *(const v4f*)(wr + 16);
        v4f q3 = half ? z : *(const v4f*)(wr + 20);     // k 700..703 -> 0
        v16h bf = cvt16(q0, q1, q2, q3);
        #pragma unroll
        for (int s = 0; s < 4; ++s) {
            const float* xr = x + (m_base + s*16 + lr)*IN_ + KMAIN_ + 8*half;
            v4f p0 = __builtin_nontemporal_load((const v4f*)(xr));
            v4f p1 = __builtin_nontemporal_load((const v4f*)(xr + 4));
            v4f p2 = __builtin_nontemporal_load((const v4f*)(xr + 16));
            v4f p3 = half ? z : __builtin_nontemporal_load((const v4f*)(xr + 20));
            v16h af = cvt16(p0, p1, p2, p3);
            c[s] = __builtin_amdgcn_wmma_f32_16x16x32_f16(
                       false, af, false, bf, (short)0, c[s], false, false);
        }
    }
    // ---- epilogue: +bias, relu, store f16 (re-read by GEMM2: keep cached)
    float bn = bias[n0 + lr];
    #pragma unroll
    for (int s = 0; s < 4; ++s) {
        #pragma unroll
        for (int r = 0; r < 8; ++r) {
            long m = m_base + s*16 + r + 8*half;
            float v = c[s][r] + bn;
            v = v > 0.f ? v : 0.f;
            hout[m*H_ + n0 + lr] = (_Float16)v;
        }
    }
}

// ---------------- GEMM2: xa = h @ A1m^T + A1_b  (f16 in, f32 out) ----------------
// A1m (32 KB f16) staged in LDS once per block; B fragments via ds_load_b128.
__global__ __launch_bounds__(256) void gemm_rec_kernel(
    const _Float16* __restrict__ hbuf, const _Float16* __restrict__ a1m,
    const float* __restrict__ bias, float* __restrict__ xa) {
    __shared__ _Float16 a1s[H_*H_];
    {
        const uint4* src = (const uint4*)a1m;
        uint4* dst = (uint4*)a1s;
        for (int i = threadIdx.x; i < (H_*H_*2)/16; i += 256) dst[i] = src[i];
    }
    __syncthreads();

    const int lane = threadIdx.x & 31;
    const int half = lane >> 4;
    const int lr   = lane & 15;
    const int n0   = (threadIdx.x >> 5) * 16;
    const long m_base = (long)blockIdx.x * 64;

    v8f c[4];
    #pragma unroll
    for (int s = 0; s < 4; ++s) c[s] = (v8f){};

    #pragma unroll
    for (int kc = 0; kc < H_; kc += 32) {
        const _Float16* br = a1s + (n0 + lr) * H_ + kc + 8*half;   // B[k][n]=A1m[n][k]
        v8h b0 = *(const v8h*)(br);
        v8h b1 = *(const v8h*)(br + 16);
        v16h bf;
        #pragma unroll
        for (int e = 0; e < 8; ++e) { bf[e] = b0[e]; bf[8+e] = b1[e]; }

        #pragma unroll
        for (int s = 0; s < 4; ++s) {
            const _Float16* ar = hbuf + (m_base + s*16 + lr) * H_ + kc;
            v8h a0 = *(const v8h*)(ar + 8*half);
            v8h a1 = *(const v8h*)(ar + 16 + 8*half);
            v16h af;
            #pragma unroll
            for (int e = 0; e < 8; ++e) { af[e] = a0[e]; af[8+e] = a1[e]; }
            c[s] = __builtin_amdgcn_wmma_f32_16x16x32_f16(
                       false, af, false, bf, (short)0, c[s], false, false);
        }
    }
    float bn = bias[n0 + lr];
    #pragma unroll
    for (int s = 0; s < 4; ++s) {
        #pragma unroll
        for (int r = 0; r < 8; ++r) {
            long m = m_base + s*16 + r + 8*half;
            xa[m*H_ + n0 + lr] = c[s][r] + bn;      // re-read by scan: keep cached
        }
    }
}

// ---------------- LIF scan over T; NT-streams histories out ----------------
__global__ void scan_kernel(const float* __restrict__ xa,
                            const float* __restrict__ mem0,
                            const float* __restrict__ thr,
                            const float* __restrict__ decay,
                            const float* __restrict__ rst,
                            float* __restrict__ out,
                            float* __restrict__ ssum_buf) {
    int tid = blockIdx.x * blockDim.x + threadIdx.x;
    if (tid >= B_*H_) return;
    int b  = tid >> 7;
    int hh = tid & (H_-1);
    float mem = mem0[tid];
    float th = thr[hh], dc = decay[hh], rs = rst[hh];
    float spk = 0.f, ss = 0.f;
    float* memh = out + O_MEM + (size_t)b*(T_+1)*H_ + hh;
    float* spkh = out + O_SPK + (size_t)b*(T_+1)*H_ + hh;
    __builtin_nontemporal_store(mem, memh);
    __builtin_nontemporal_store(0.f, spkh);
    const float* xr = xa + (size_t)b*T_*H_ + hh;
    for (int t = 0; t < T_; ++t) {
        float xv = xr[(size_t)t*H_];
        mem = rs*spk + mem*dc*(1.f - spk) + xv;
        spk = (mem - th) > 0.f ? 1.f : 0.f;
        ss += spk;
        __builtin_nontemporal_store(mem, memh + (size_t)(t+1)*H_);
        __builtin_nontemporal_store(spk, spkh + (size_t)(t+1)*H_);
    }
    ssum_buf[tid] = ss;
}

// ---------------- out = (ssum/T) @ fc_out_w^T + b ----------------
__global__ void fcout_kernel(const float* __restrict__ ssum,
                             const float* __restrict__ w,
                             const float* __restrict__ bias,
                             float* __restrict__ out) {
    int tid = blockIdx.x * blockDim.x + threadIdx.x;
    if (tid >= B_*OUT_) return;
    int b = tid / OUT_, o = tid % OUT_;
    float acc = bias[o];
    const float* sr = ssum + (size_t)b*H_;
    const float* wr = w + (size_t)o*H_;
    const float inv_t = 1.f / (float)T_;
    #pragma unroll 4
    for (int h = 0; h < H_; ++h) acc += (sr[h] * inv_t) * wr[h];
    out[tid] = acc;
}

extern "C" void kernel_launch(void* const* d_in, const int* in_sizes, int n_in,
                              void* d_out, int out_size, void* d_ws, size_t ws_size,
                              hipStream_t stream) {
    (void)in_sizes; (void)n_in; (void)out_size; (void)ws_size;
    const float* x        = (const float*)d_in[0];
    const int*   A1_mask  = (const int*)  d_in[1];
    const float* fc_in_w  = (const float*)d_in[2];
    const float* fc_in_b  = (const float*)d_in[3];
    const float* A1_w     = (const float*)d_in[4];
    const float* A1_b     = (const float*)d_in[5];
    const float* fc_out_w = (const float*)d_in[6];
    const float* fc_out_b = (const float*)d_in[7];
    const float* thr      = (const float*)d_in[8];
    const float* decay    = (const float*)d_in[9];
    const float* rst      = (const float*)d_in[10];
    const float* mem0     = (const float*)d_in[11];

    char* ws = (char*)d_ws;
    _Float16* a1m  = (_Float16*)(ws + WS_A1M);
    _Float16* hbuf = (_Float16*)(ws + WS_H);
    float*    xaf  = (float*)   (ws + WS_XA);
    float*    ssum = (float*)   (ws + WS_SSUM);
    float*    out  = (float*)d_out;

    prep_kernel<<<1, 256, 0, stream>>>(A1_w, A1_mask, a1m, out + O_ANRM);
    gemm_in_kernel<<<BT_/64, 256, 0, stream>>>(x, fc_in_w, fc_in_b, hbuf);
    gemm_rec_kernel<<<BT_/64, 256, 0, stream>>>(hbuf, a1m, A1_b, xaf);
    scan_kernel<<<(B_*H_)/256, 256, 0, stream>>>(xaf, mem0, thr, decay, rst, out, ssum);
    fcout_kernel<<<(B_*OUT_ + 255)/256, 256, 0, stream>>>(ssum, fc_out_w, fc_out_b, out + O_OUT);
}
